// AttentionBlock_59107339928233
// MI455X (gfx1250) — matrix operand order repaired and checked by reading, not compiled, not run on hardware
//
#include <hip/hip_runtime.h>
#include <cstdint>

#define EMBED  512
#define NHEADS 8
#define HDIM   64
#define BATCH  4
#define SEQ    2048
#define MTOK   (BATCH * SEQ)   // 8192 tokens

// scores are produced pre-multiplied by (1/sqrt(D)) * log2(e) so softmax is
// pure exp2 in the inner loop: fold the constant into Q at projection time.
#define QSCALE 0.18033688011112042f   // 0.125 * 1.4426950408889634

typedef __attribute__((ext_vector_type(16))) __bf16         v16bf;
typedef __attribute__((ext_vector_type(8)))  __bf16         v8bf;
typedef __attribute__((ext_vector_type(8)))  float          v8f;
typedef __attribute__((ext_vector_type(16))) unsigned short v16us;
typedef __attribute__((ext_vector_type(4)))  unsigned int   v4u;
typedef __attribute__((ext_vector_type(8)))  int            v8i;
typedef __attribute__((ext_vector_type(4)))  int            v4i;

// ---------- helpers ----------

__device__ __forceinline__ uint16_t f2bf(float f) {
    union { float f; uint32_t u; } c; c.f = f;
    uint32_t u = c.u;
    uint32_t r = (u + 0x7fffu + ((u >> 16) & 1u)) >> 16;   // RNE bf16
    return (uint16_t)r;
}

__device__ __forceinline__ float gelu_exact(float v) {
    return 0.5f * v * (1.0f + erff(v * 0.70710678118654752f));
}

// A-fragment (16x32 bf16, MxK), row-major source, leading dim ld.
// lanes 0-15: row M=lane, K {0..7,16..23}; lanes 16-31: row M=lane-16, K {8..15,24..31}.
__device__ __forceinline__ v16bf load_afrag(const uint16_t* base, int ld,
                                            int row0, int k0, int lane) {
    const int r   = row0 + (lane & 15);
    const int hi8 = (lane >> 4) & 1;
    const uint16_t* p = base + (size_t)r * ld + k0 + hi8 * 8;
    v8bf lo = *(const v8bf*)(p);
    v8bf hi = *(const v8bf*)(p + 16);
    return __builtin_shufflevector(lo, hi, 0,1,2,3,4,5,6,7,8,9,10,11,12,13,14,15);
}

// B-fragment (32x16 bf16, KxN): column n of B = row n of row-major [N,K] (ld).
// lane = column; lanes 0-15 hold K {0..7,16..23}, lanes 16-31 hold K {8..15,24..31}.
__device__ __forceinline__ v16bf load_bfrag(const uint16_t* base, int ld,
                                            int col0, int k0, int lane) {
    const int c    = col0 + (lane & 15);
    const int hi8  = ((lane >> 4) & 1) * 8;
    const uint16_t* p = base + (size_t)c * ld + k0 + hi8;
    v8bf lo = *(const v8bf*)(p);
    v8bf hi = *(const v8bf*)(p + 16);
    return __builtin_shufflevector(lo, hi, 0,1,2,3,4,5,6,7,8,9,10,11,12,13,14,15);
}

__device__ __forceinline__ v8f wmma_bf16(v16bf a, v16bf b, v8f c) {
    return __builtin_amdgcn_wmma_f32_16x16x32_bf16(false, a, false, b,
                                                   (short)0, c, false, false);
}

// Issue a TDM 2D tile load (bf16 elements) into LDS.
// D# layout per CDNA5 ISA §8: group0 = {count, lds_addr, global_addr, type=2},
// group1 = {data_size, tensor dims, tile dims, dim0 stride}. 2D tile ->
// groups 2/3 unused (zero).
__device__ __forceinline__ void tdm_load_2d_bf16(uint32_t lds_byte_addr,
                                                 const uint16_t* gsrc,
                                                 uint32_t tensor_d0, uint32_t tensor_d1,
                                                 uint32_t stride0_elems,
                                                 uint32_t tile_d0, uint32_t tile_d1) {
    const uint64_t ga = (uint64_t)(uintptr_t)gsrc;
    v4u g0;
    g0[0] = 1u;                                                  // count=1 (user D#)
    g0[1] = lds_byte_addr;                                       // LDS dest (bytes)
    g0[2] = (uint32_t)ga;                                        // global_addr[31:0]
    g0[3] = (uint32_t)((ga >> 32) & 0x01FFFFFFu) | (2u << 30);   // addr[56:32] | type=2
    v8i g1;
    g1[0] = 0x00010000;                                          // data_size=1 (2B)
    g1[1] = (int)((tensor_d0 & 0xFFFFu) << 16);                  // tensor_dim0[15:0]
    g1[2] = (int)(((tensor_d0 >> 16) & 0xFFFFu) |
                  ((tensor_d1 & 0xFFFFu) << 16));                // dim0 hi | dim1 lo
    g1[3] = (int)(((tensor_d1 >> 16) & 0xFFFFu) |
                  ((tile_d0 & 0xFFFFu) << 16));                  // dim1 hi | tile_dim0
    g1[4] = (int)(tile_d1 & 0xFFFFu);                            // tile_dim1 (tile_dim2=0)
    g1[5] = (int)stride0_elems;                                  // tensor_dim0_stride lo32
    g1[6] = 0;
    g1[7] = 0;
    v4i z4 = {0, 0, 0, 0};
#if defined(__clang_major__) && (__clang_major__ >= 23)
    v8i z8 = {0, 0, 0, 0, 0, 0, 0, 0};
    __builtin_amdgcn_tensor_load_to_lds(g0, g1, z4, z4, z8, 0);
#else
    __builtin_amdgcn_tensor_load_to_lds(g0, g1, z4, z4, 0);
#endif
}

__device__ __forceinline__ uint32_t lds_addr_of(const void* p) {
    return (uint32_t)(uintptr_t)p;   // LDS aperture: low 32 bits = LDS byte offset
}

// ---------- kernels ----------

__global__ void cvt_bf16_kernel(const float* __restrict__ in,
                                uint16_t* __restrict__ out, int n) {
    int i = blockIdx.x * blockDim.x + threadIdx.x;
    if (i < n) out[i] = f2bf(in[i]);
}

// y = gelu(A[M,K] @ W[N,K]^T + bias[N]); each wave computes a 16x64 strip.
// MODE 0: bf16 -> [b,h,s,d]            (K buffer)
// MODE 1: bf16 -> [b,h,d,s]            (V transposed buffer)
// MODE 2: fp32 -> [m, EMBED]           (final output)
// MODE 3: bf16 * QSCALE -> [b,h,s,d]   (Q buffer, softmax scale folded in)
template <int MODE>
__global__ __launch_bounds__(256)
void proj_kernel(const uint16_t* __restrict__ A, const uint16_t* __restrict__ W,
                 const float* __restrict__ bias, void* __restrict__ outp) {
    const int lane = threadIdx.x & 31;
    const int wave = threadIdx.x >> 5;
    const int n0 = blockIdx.x * 64;
    const int m0 = (blockIdx.y * 8 + wave) * 16;

    v8f acc0 = {}, acc1 = {}, acc2 = {}, acc3 = {};
    #pragma unroll 2
    for (int k0 = 0; k0 < EMBED; k0 += 32) {
        v16bf a = load_afrag(A, EMBED, m0, k0, lane);
        acc0 = wmma_bf16(a, load_bfrag(W, EMBED, n0,      k0, lane), acc0);
        acc1 = wmma_bf16(a, load_bfrag(W, EMBED, n0 + 16, k0, lane), acc1);
        acc2 = wmma_bf16(a, load_bfrag(W, EMBED, n0 + 32, k0, lane), acc2);
        acc3 = wmma_bf16(a, load_bfrag(W, EMBED, n0 + 48, k0, lane), acc3);
    }

    const int col = lane & 15;
    const int hi8 = (lane >> 4) & 1;
    v8f* accs[4] = { &acc0, &acc1, &acc2, &acc3 };
    #pragma unroll
    for (int t = 0; t < 4; ++t) {
        const int f = n0 + t * 16 + col;
        const float bv = bias[f];
        const int h = f / HDIM, d = f % HDIM;
        #pragma unroll
        for (int r = 0; r < 8; ++r) {
            const int m = m0 + r + hi8 * 8;
            float g = gelu_exact((*accs[t])[r] + bv);
            if (MODE == 2) {
                ((float*)outp)[(size_t)m * EMBED + f] = g;
            } else {
                if (MODE == 3) g *= QSCALE;
                const uint16_t gb = f2bf(g);
                const int b = m / SEQ, s = m % SEQ;
                if (MODE == 1)
                    ((uint16_t*)outp)[((size_t)(b * NHEADS + h) * HDIM + d) * SEQ + s] = gb;
                else
                    ((uint16_t*)outp)[((size_t)(b * NHEADS + h) * SEQ + s) * HDIM + d] = gb;
            }
        }
    }
}

// Flash attention with TDM-staged K/V tiles.
// Per block: one (b,h), 128 query rows (8 waves x 16). Per 64-key step, wave 0
// issues two tensor_load_to_lds DMAs (64x64 K tile, 64x64 V tile) into the
// next LDS buffer, waits TENSORcnt<=2 (gating only the in-flight pair), and
// all waves consume the current buffer via ds loads. Q pre-scaled so scores
// are in the exp2 domain; running row-sum l rides the matrix pipe as P @ 1.
__global__ __launch_bounds__(256)
void attn_kernel(const uint16_t* __restrict__ qbf, const uint16_t* __restrict__ kbf,
                 const uint16_t* __restrict__ vtbf, uint16_t* __restrict__ obf) {
    // [buf][0]=K tile (rows j, cols d), [buf][1]=V tile (rows d, cols j)
    __shared__ __attribute__((aligned(64))) uint16_t kv_lds[2][2][64 * 64];  // 32 KB
    __shared__ __attribute__((aligned(64))) uint16_t p_lds_all[8 * 16 * 64]; // 16 KB

    const int lane = threadIdx.x & 31;
    const int wave = threadIdx.x >> 5;
    const int bh = blockIdx.x;
    const int i0 = blockIdx.y * 128 + wave * 16;

    const uint16_t* qb = qbf  + (size_t)bh * SEQ * HDIM;
    const uint16_t* kb = kbf  + (size_t)bh * SEQ * HDIM;   // [S, D]
    const uint16_t* vb = vtbf + (size_t)bh * HDIM * SEQ;   // [D, S]
    uint16_t* plds = p_lds_all + wave * (16 * 64);

    const v16bf qa0 = load_afrag(qb, HDIM, i0, 0,  lane);
    const v16bf qa1 = load_afrag(qb, HDIM, i0, 32, lane);

    v16bf vones;   // all-ones B matrix: row-sum via WMMA
    { union { v16us u; v16bf b; } c; c.u = (v16us)(unsigned short)0x3f80; vones = c.b; }

    v8f o0 = {}, o1 = {}, o2 = {}, o3 = {}, o4 = {};   // o4 = running l
    float mrow[8];
    #pragma unroll
    for (int r = 0; r < 8; ++r) mrow[r] = -1e30f;

    const int col = lane & 15;
    const int hi8 = (lane >> 4) & 1;
    const int nsteps = SEQ / 64;

    if (wave == 0) {   // prologue: DMA step-0 tiles into buffer 0
        tdm_load_2d_bf16(lds_addr_of(&kv_lds[0][0][0]), kb, HDIM, SEQ, HDIM, 64, 64);
        tdm_load_2d_bf16(lds_addr_of(&kv_lds[0][1][0]), vb, SEQ, HDIM, SEQ, 64, 64);
    }

    for (int t = 0; t < nsteps; ++t) {
        const int cur = t & 1;
        const int j0 = t * 64;
        if (wave == 0) {
            if (t + 1 < nsteps) {
                const int jn = j0 + 64;
                tdm_load_2d_bf16(lds_addr_of(&kv_lds[cur ^ 1][0][0]),
                                 kb + (size_t)jn * HDIM, HDIM, SEQ, HDIM, 64, 64);
                tdm_load_2d_bf16(lds_addr_of(&kv_lds[cur ^ 1][1][0]),
                                 vb + jn, SEQ, HDIM, SEQ, 64, 64);
                __builtin_amdgcn_s_wait_tensorcnt(2);   // current buffer complete
            } else {
                __builtin_amdgcn_s_wait_tensorcnt(0);
            }
        }
        __syncthreads();   // publish current K/V tiles to all waves

        const uint16_t* ldsK = &kv_lds[cur][0][0];
        const uint16_t* ldsV = &kv_lds[cur][1][0];

        // ---- scores: four 16x16 tiles covering 64 keys (from LDS) ----
        v8f s0 = {}, s1 = {}, s2 = {}, s3 = {};
        s0 = wmma_bf16(qa0, load_bfrag(ldsK, HDIM, 0,  0,  lane), s0);
        s0 = wmma_bf16(qa1, load_bfrag(ldsK, HDIM, 0,  32, lane), s0);
        s1 = wmma_bf16(qa0, load_bfrag(ldsK, HDIM, 16, 0,  lane), s1);
        s1 = wmma_bf16(qa1, load_bfrag(ldsK, HDIM, 16, 32, lane), s1);
        s2 = wmma_bf16(qa0, load_bfrag(ldsK, HDIM, 32, 0,  lane), s2);
        s2 = wmma_bf16(qa1, load_bfrag(ldsK, HDIM, 32, 32, lane), s2);
        s3 = wmma_bf16(qa0, load_bfrag(ldsK, HDIM, 48, 0,  lane), s3);
        s3 = wmma_bf16(qa1, load_bfrag(ldsK, HDIM, 48, 32, lane), s3);

        // ---- online softmax: single merged max tree per row ----
        float corr[8];
        #pragma unroll
        for (int r = 0; r < 8; ++r) {
            float tm = fmaxf(fmaxf(s0[r], s1[r]), fmaxf(s2[r], s3[r]));
            #pragma unroll
            for (int mk = 1; mk < 16; mk <<= 1)
                tm = fmaxf(tm, __shfl_xor(tm, mk, 32));
            const float mnew = fmaxf(mrow[r], tm);
            corr[r] = exp2f(mrow[r] - mnew);
            mrow[r] = mnew;
            s0[r] = exp2f(s0[r] - mnew);
            s1[r] = exp2f(s1[r] - mnew);
            s2[r] = exp2f(s2[r] - mnew);
            s3[r] = exp2f(s3[r] - mnew);
        }
        #pragma unroll
        for (int r = 0; r < 8; ++r) {
            o0[r] *= corr[r]; o1[r] *= corr[r]; o2[r] *= corr[r];
            o3[r] *= corr[r]; o4[r] *= corr[r];
        }

        // ---- re-layout P (C layout fp32 -> A fragments bf16) via wave LDS ----
        #pragma unroll
        for (int r = 0; r < 8; ++r) {
            const int row = r + hi8 * 8;
            uint16_t* prow = plds + row * 64 + col;
            prow[0]  = f2bf(s0[r]);
            prow[16] = f2bf(s1[r]);
            prow[32] = f2bf(s2[r]);
            prow[48] = f2bf(s3[r]);
        }
        v16bf pA0, pA1;
        {
            const int row = lane & 15;
            const uint16_t* p0 = plds + row * 64 + hi8 * 8;
            v8bf lo = *(const v8bf*)(p0);
            v8bf hi = *(const v8bf*)(p0 + 16);
            pA0 = __builtin_shufflevector(lo, hi, 0,1,2,3,4,5,6,7,8,9,10,11,12,13,14,15);
            const uint16_t* p1 = p0 + 32;
            lo = *(const v8bf*)(p1);
            hi = *(const v8bf*)(p1 + 16);
            pA1 = __builtin_shufflevector(lo, hi, 0,1,2,3,4,5,6,7,8,9,10,11,12,13,14,15);
        }

        // ---- O += P @ V (K=64, from LDS), l += P @ 1 on the matrix pipe ----
        o0 = wmma_bf16(pA0, load_bfrag(ldsV, 64, 0,  0,  lane), o0);
        o0 = wmma_bf16(pA1, load_bfrag(ldsV, 64, 0,  32, lane), o0);
        o1 = wmma_bf16(pA0, load_bfrag(ldsV, 64, 16, 0,  lane), o1);
        o1 = wmma_bf16(pA1, load_bfrag(ldsV, 64, 16, 32, lane), o1);
        o2 = wmma_bf16(pA0, load_bfrag(ldsV, 64, 32, 0,  lane), o2);
        o2 = wmma_bf16(pA1, load_bfrag(ldsV, 64, 32, 32, lane), o2);
        o3 = wmma_bf16(pA0, load_bfrag(ldsV, 64, 48, 0,  lane), o3);
        o3 = wmma_bf16(pA1, load_bfrag(ldsV, 64, 48, 32, lane), o3);
        o4 = wmma_bf16(pA0, vones, o4);
        o4 = wmma_bf16(pA1, vones, o4);

        __syncthreads();   // all waves done with buffer before it is re-filled
    }

    // ---- finalize: divide by l (lane-local in o4), scatter to [b,s,EMBED] ----
    const int b = bh / NHEADS, h = bh % NHEADS;
    #pragma unroll
    for (int r = 0; r < 8; ++r) {
        const int row = r + hi8 * 8;
        const int s = i0 + row;
        const float inv = 1.0f / o4[r];
        uint16_t* orow = obf + ((size_t)(b * SEQ + s) * EMBED) + h * HDIM;
        orow[0  + col] = f2bf(o0[r] * inv);
        orow[16 + col] = f2bf(o1[r] * inv);
        orow[32 + col] = f2bf(o2[r] * inv);
        orow[48 + col] = f2bf(o3[r] * inv);
    }
}

// ---------- host launch ----------

extern "C" void kernel_launch(void* const* d_in, const int* in_sizes, int n_in,
                              void* d_out, int out_size, void* d_ws, size_t ws_size,
                              hipStream_t stream) {
    const float* x  = (const float*)d_in[0];
    const float* Wq = (const float*)d_in[1];
    const float* bq = (const float*)d_in[2];
    const float* Wk = (const float*)d_in[3];
    const float* bk = (const float*)d_in[4];
    const float* Wv = (const float*)d_in[5];
    const float* bv = (const float*)d_in[6];
    const float* Wo = (const float*)d_in[7];
    const float* bo = (const float*)d_in[8];
    float* out = (float*)d_out;

    uint8_t* ws = (uint8_t*)d_ws;
    const size_t SZ_X = (size_t)MTOK * EMBED * 2;                 // 8 MB
    const size_t SZ_W = (size_t)EMBED * EMBED * 2;                // 0.5 MB
    const size_t SZ_H = (size_t)BATCH * NHEADS * SEQ * HDIM * 2;  // 8 MB

    uint16_t* xbf  = (uint16_t*)(ws);
    uint16_t* wqb  = (uint16_t*)(ws + SZ_X);
    uint16_t* wkb  = (uint16_t*)(ws + SZ_X + SZ_W);
    uint16_t* wvb  = (uint16_t*)(ws + SZ_X + 2 * SZ_W);
    uint16_t* wob  = (uint16_t*)(ws + SZ_X + 3 * SZ_W);
    uint16_t* qbf  = (uint16_t*)(ws + SZ_X + 4 * SZ_W);
    uint16_t* kbf  = (uint16_t*)(ws + SZ_X + 4 * SZ_W + SZ_H);
    uint16_t* vtbf = (uint16_t*)(ws + SZ_X + 4 * SZ_W + 2 * SZ_H);
    uint16_t* obf  = (uint16_t*)(ws + SZ_X + 4 * SZ_W + 3 * SZ_H);

    {   // fp32 -> bf16
        const int nx = MTOK * EMBED;
        cvt_bf16_kernel<<<(nx + 255) / 256, 256, 0, stream>>>(x, xbf, nx);
        const int nw = EMBED * EMBED;
        cvt_bf16_kernel<<<(nw + 255) / 256, 256, 0, stream>>>(Wq, wqb, nw);
        cvt_bf16_kernel<<<(nw + 255) / 256, 256, 0, stream>>>(Wk, wkb, nw);
        cvt_bf16_kernel<<<(nw + 255) / 256, 256, 0, stream>>>(Wv, wvb, nw);
        cvt_bf16_kernel<<<(nw + 255) / 256, 256, 0, stream>>>(Wo, wob, nw);
    }

    {   // projections: Q (scaled), K, V (transposed)
        dim3 grid(EMBED / 64, MTOK / 128);
        proj_kernel<3><<<grid, 256, 0, stream>>>(xbf, wqb, bq, qbf);
        proj_kernel<0><<<grid, 256, 0, stream>>>(xbf, wkb, bk, kbf);
        proj_kernel<1><<<grid, 256, 0, stream>>>(xbf, wvb, bv, vtbf);
    }

    {   // flash attention (TDM-staged K/V) -> obf [b,s,EMBED] bf16
        dim3 grid(BATCH * NHEADS, SEQ / 128);
        attn_kernel<<<grid, 256, 0, stream>>>(qbf, kbf, vtbf, obf);
    }

    {   // output projection -> fp32 d_out
        dim3 grid(EMBED / 64, MTOK / 128);
        proj_kernel<2><<<grid, 256, 0, stream>>>(obf, wob, bo, out);
    }

    (void)in_sizes; (void)n_in; (void)out_size; (void)ws_size;
}
